// DinoAdapter_9569187135811
// MI455X (gfx1250) — matrix operand order repaired and checked by reading, not compiled, not run on hardware
//
#include <hip/hip_runtime.h>

typedef __attribute__((ext_vector_type(16))) _Float16 v16h;
typedef __attribute__((ext_vector_type(8)))  _Float16 v8h;
typedef __attribute__((ext_vector_type(4)))  _Float16 v4h;
typedef __attribute__((ext_vector_type(2)))  _Float16 v2h;
typedef __attribute__((ext_vector_type(8)))  float    v8f;
typedef __attribute__((ext_vector_type(4)))  unsigned int v4u;
typedef __attribute__((ext_vector_type(8)))  int      v8i;
typedef __attribute__((ext_vector_type(4)))  int      v4i;

#define DMODEL 768
#define NHEAD  12
#define NLVL   3
#define NPTS   4
#define HDIM   64
#define NBATCH 4
#define LQTOK  4096
#define LINTOK 16464
#define QWGRID 64
#define QHGRID 64
#define NCAT   432   // 288 offset cols + 144 attn cols

// ---------------- workspace layout (bytes, all 256-aligned) ----------------
#define OFF_FEATN  ((size_t)0)                         // f16 (65856,768)  101,154,816
#define OFF_QN     (OFF_FEATN + 101154816ull)          // f16 (16384,768)   25,165,824
#define OFF_VALUE  (OFF_QN    + 25165824ull)           // f16 (65856,768)  101,154,816
#define OFF_WVALP  (OFF_VALUE + 101154816ull)          // f16 packed         1,179,648
#define OFF_WCATF  (OFF_WVALP + 1179648ull)            // f32 (768,432)      1,327,104
#define OFF_WCATP  (OFF_WCATF + 1327104ull)            // f16 packed           663,552
#define OFF_WOUTP  (OFF_WCATP + 663552ull)             // f16 packed         1,179,648
#define OFF_BCAT   (OFF_WOUTP + 1179648ull)            // f32 (432)              2,048
#define OFF_OFFAW  (OFF_BCAT  + 2048ull)               // f32 (16384,432)   28,311,552
#define OFF_AW     (OFF_OFFAW + 28311552ull)           // f32 (16384,12,12)  9,437,184
#define OFF_ATTN   (OFF_AW    + 9437184ull)            // f16 (16384,768)   25,165,824

// ---------------- weight prep ----------------
__global__ void concat_w_kernel(const float* __restrict__ Woff,
                                const float* __restrict__ Wattn,
                                float* __restrict__ Wcat) {
  int t = blockIdx.x * blockDim.x + threadIdx.x;
  if (t >= DMODEL * NCAT) return;
  int k = t / NCAT, n = t % NCAT;
  Wcat[t] = (n < 288) ? Woff[k * 288 + n] : Wattn[k * 144 + (n - 288)];
}

__global__ void concat_b_kernel(const float* __restrict__ boff,
                                const float* __restrict__ battn,
                                float* __restrict__ bcat) {
  int t = blockIdx.x * blockDim.x + threadIdx.x;
  if (t < 288) bcat[t] = boff[t];
  else if (t < NCAT) bcat[t] = battn[t - 288];
}

// Pack f32 row-major weight (K x N) into WMMA B-operand tiles (f16).
// Per ISA 7.12.2 (32x16 f16 B): lanes 0-15 hold col N=lane, K=kt*32+{0..7,16..23};
// lanes 16-31 hold col N=lane-16, K=kt*32+{8..15,24..31}. Each lane's 16 values are
// stored contiguously so the GEMM reads one aligned 32B chunk per lane.
__global__ void pack_w_kernel(const float* __restrict__ W, _Float16* __restrict__ P,
                              int K, int N) {
  int t = blockIdx.x * blockDim.x + threadIdx.x;
  int KT = K >> 5, NT = N >> 4;
  if (t >= KT * NT * 32) return;
  int lane = t & 31;
  int tile = t >> 5;               // tile = nt*KT + kt  (matches GEMM indexing)
  int kt = tile % KT, nt = tile / KT;
  int n  = (nt << 4) + (lane & 15);
  int k0 = (kt << 5) + ((lane >> 4) << 3);
  _Float16* dst = P + ((size_t)tile * 32 + lane) * 16;
#pragma unroll
  for (int j = 0; j < 16; ++j)
    dst[j] = (_Float16)W[(size_t)(k0 + j + ((j >> 3) << 3)) * N + n];
}

// ---------------- layernorm + cast to f16 (one wave per 768-wide row) ------
__global__ void ln_cast_kernel(const float* __restrict__ src,
                               const float* __restrict__ g,
                               const float* __restrict__ b,
                               _Float16* __restrict__ dst, int rows) {
  int wave = (int)((blockIdx.x * (size_t)blockDim.x + threadIdx.x) >> 5);
  int lane = threadIdx.x & 31;
  if (wave >= rows) return;
  const float* r = src + (size_t)wave * DMODEL;
  float4 v[6];
  float s = 0.f;
#pragma unroll
  for (int j = 0; j < 6; ++j) {
    v[j] = *(const float4*)(r + j * 128 + lane * 4);
    s += v[j].x + v[j].y + v[j].z + v[j].w;
  }
#pragma unroll
  for (int o = 16; o; o >>= 1) s += __shfl_xor(s, o, 32);
  float mean = s * (1.f / 768.f);
  float vs = 0.f;
#pragma unroll
  for (int j = 0; j < 6; ++j) {
    v[j].x -= mean; v[j].y -= mean; v[j].z -= mean; v[j].w -= mean;
    vs += v[j].x * v[j].x + v[j].y * v[j].y + v[j].z * v[j].z + v[j].w * v[j].w;
  }
#pragma unroll
  for (int o = 16; o; o >>= 1) vs += __shfl_xor(vs, o, 32);
  float inv = rsqrtf(vs * (1.f / 768.f) + 1e-6f);
  _Float16* o16 = dst + (size_t)wave * DMODEL;
#pragma unroll
  for (int j = 0; j < 6; ++j) {
    int c = j * 128 + lane * 4;
    float4 gg = *(const float4*)(g + c);
    float4 bb = *(const float4*)(b + c);
    v4h h;
    h.x = (_Float16)(v[j].x * inv * gg.x + bb.x);
    h.y = (_Float16)(v[j].y * inv * gg.y + bb.y);
    h.z = (_Float16)(v[j].z * inv * gg.z + bb.z);
    h.w = (_Float16)(v[j].w * inv * gg.w + bb.w);
    *(v4h*)(o16 + c) = h;
  }
}

// ---------------- WMMA GEMM with TDM-staged A panel ------------------------
// One block = 4 waves sharing one 16-row M tile. A 16xK f16 panel (24KB,
// contiguous in row-major A) is DMA'd into LDS by the Tensor Data Mover
// (tracked by TENSORcnt), then each wave computes a 16 x (16*NTW) strip.
// MODE 0: out f16 = acc + bias          (value projection)
// MODE 1: out f32 = acc + bias          (offset/attn projection)
// MODE 2: out f32 = resid + gamma[n]*(acc + bias)   (output projection)
#define HAVE_TDM (__has_builtin(__builtin_amdgcn_tensor_load_to_lds) && \
                  __has_builtin(__builtin_amdgcn_s_wait_tensorcnt))

template <int MODE, int NTW>
__global__ void gemm_tdm_kernel(const _Float16* __restrict__ A,
                                const _Float16* __restrict__ Bp,
                                const float* __restrict__ bias,
                                void* __restrict__ out,
                                int M, int N, int K,
                                const float* __restrict__ resid,
                                const float* __restrict__ gamma) {
  __shared__ _Float16 sA[16 * DMODEL];   // 24 KB, only LDS object -> offset 0
  int KT = K >> 5;
  int stripsN = N / (16 * NTW);
  int sgN = (stripsN + 3) >> 2;          // strip-groups per M tile (4 waves each)
  int tm = blockIdx.x / sgN;
  int sg = blockIdx.x % sgN;
  int waveid = threadIdx.x >> 5;
  int lane = threadIdx.x & 31;
  int sn = sg * 4 + waveid;
  int lrow = lane & 15, half = lane >> 4;

#if HAVE_TDM
  if (waveid == 0) {
    // Tensor DMA descriptor (ISA 08 "Tensor DMA Descriptor"):
    // 2D tile: tile_dim0 = K elements (f16), tile_dim1 = 16 rows,
    // tensor_dim0_stride = K, destination LDS offset 0.
    unsigned long long ga =
        (unsigned long long)(size_t)(A + (size_t)(tm << 4) * K);
    v4u g0 = { 1u,                                    // count=1 (valid), user desc
               0u,                                    // lds_addr = 0
               (unsigned)ga,                          // global_addr[31:0]
               (unsigned)(ga >> 32) | (2u << 30) };   // global_addr[56:32] | type=2
    v8i g1 = { 1 << 16,                               // data_size = 2 bytes
               (int)((K & 0xffff) << 16),             // tensor_dim0[15:0]
               (int)((K >> 16) | (16 << 16)),         // tensor_dim0[31:16] | tensor_dim1=16
               (int)(K << 16),                        // tile_dim0 = K
               16,                                    // tile_dim1 = 16 (tile_dim2 = 0)
               K,                                     // tensor_dim0_stride[31:0]
               0, 0 };                                // stride hi / dim1_stride = 0
    v4i z4 = { 0, 0, 0, 0 };
#if defined(__clang_major__) && (__clang_major__ >= 23)
    v8i z8 = { 0, 0, 0, 0, 0, 0, 0, 0 };
    __builtin_amdgcn_tensor_load_to_lds(g0, g1, z4, z4, z8, 0);
#else
    __builtin_amdgcn_tensor_load_to_lds(g0, g1, z4, z4, 0);
#endif
  }
  __builtin_amdgcn_s_wait_tensorcnt(0);
#else
  {
    const v8h* src = (const v8h*)(A + (size_t)(tm << 4) * K);
    v8h* dst = (v8h*)sA;
    for (int i = threadIdx.x; i < (16 * DMODEL) / 8; i += 128) dst[i] = src[i];
  }
#endif
  __syncthreads();

  v8f acc[NTW];
#pragma unroll
  for (int i = 0; i < NTW; ++i) acc[i] = (v8f){};

  bool active = sn < stripsN;
  if (active) {
    // A operand per ISA 7.12.2 (16x32 f16): lanes 0-15 row M=lane hold
    // K {0..7,16..23}; lanes 16-31 row M=lane-16 hold K {8..15,24..31}.
    const _Float16* aL = sA + (size_t)lrow * K + (half << 3);
    for (int kt = 0; kt < KT; ++kt) {
      union { v16h v; v8h h[2]; } ua;
      ua.h[0] = *(const v8h*)(aL + (kt << 5));
      ua.h[1] = *(const v8h*)(aL + (kt << 5) + 16);
#pragma unroll
      for (int i = 0; i < NTW; ++i) {
        const v16h* bptr =
            (const v16h*)(Bp + (((size_t)(sn * NTW + i) * KT + kt) * 32 + lane) * 16);
        v16h bv = *bptr;
        acc[i] = __builtin_amdgcn_wmma_f32_16x16x32_f16(
            false, ua.v, false, bv, (short)0, acc[i], false, false);
      }
    }
    // C/D layout: lane<16 -> (M=r, N=lane); lane>=16 -> (M=r+8, N=lane-16)
    int mbase = (tm << 4) + (half << 3);
#pragma unroll
    for (int i = 0; i < NTW; ++i) {
      int n = ((sn * NTW + i) << 4) + lrow;
      float bn = bias[n];
#pragma unroll
      for (int r = 0; r < 8; ++r) {
        float vv = acc[i][r] + bn;
        size_t idx = (size_t)(mbase + r) * N + n;
        if (MODE == 0)      ((_Float16*)out)[idx] = (_Float16)vv;
        else if (MODE == 1) ((float*)out)[idx] = vv;
        else                ((float*)out)[idx] = resid[idx] + gamma[n] * vv;
      }
    }
  }
}

// ---------------- softmax over the 12 attention logits ---------------------
__global__ void softmax_kernel(const float* __restrict__ offaw, float* __restrict__ aw) {
  int t = blockIdx.x * blockDim.x + threadIdx.x;
  if (t >= NBATCH * LQTOK * NHEAD) return;
  int h = t % NHEAD;
  int row = t / NHEAD;                           // b*LQ + lq
  const float* src = offaw + (size_t)row * NCAT + 288 + h * 12;
  float v[12], m = -3.4e38f;
#pragma unroll
  for (int j = 0; j < 12; ++j) { v[j] = src[j]; m = fmaxf(m, v[j]); }
  float s = 0.f;
#pragma unroll
  for (int j = 0; j < 12; ++j) { v[j] = __expf(v[j] - m); s += v[j]; }
  float inv = 1.f / s;
  float* d = aw + (size_t)t * 12;
#pragma unroll
  for (int j = 0; j < 12; ++j) d[j] = v[j] * inv;
}

// ---------------- deformable sampling: one wave per (b, lq, head) ----------
// Each lane owns 2 consecutive channels (4B f16x2 gathers). The f16 value
// tensor (101 MB) fits the 192 MB L2, halving gather bytes vs f32. f32
// accumulate, f16 output feeding the out-projection WMMA GEMM.
__global__ void sample_kernel(const _Float16* __restrict__ value,
                              const float* __restrict__ offaw,
                              const float* __restrict__ aw,
                              _Float16* __restrict__ attn) {
  int wave = (int)((blockIdx.x * (size_t)blockDim.x + threadIdx.x) >> 5);
  int lane = threadIdx.x & 31;
  if (wave >= NBATCH * LQTOK * NHEAD) return;
  int h = wave % NHEAD;
  int row = wave / NHEAD;                        // b*LQ + lq
  int lq = row % LQTOK, b = row / LQTOK;
  int ch = lane << 1;

  float rx = ((lq % QWGRID) + 0.5f) * (1.f / QWGRID);
  float ry = ((lq / QWGRID) + 0.5f) * (1.f / QHGRID);
  const float* offrow = offaw + (size_t)row * NCAT;
  const float* awrow  = aw + (size_t)wave * 12;

  const int HL[3] = {112, 56, 28};
  const int ST[3] = {0, 12544, 15680};

  float ax = 0.f, ay = 0.f;
#pragma unroll
  for (int l = 0; l < NLVL; ++l) {
    int hl = HL[l], wl = HL[l], st = ST[l];
    float fw = (float)wl, fh = (float)hl;
#pragma unroll
    for (int p = 0; p < NPTS; ++p) {
      int oc = ((h * NLVL + l) * NPTS + p) * 2;
      float ox = offrow[oc], oy = offrow[oc + 1];
      float a  = awrow[l * NPTS + p];
      float x = (rx + ox / fw) * fw - 0.5f;
      float y = (ry + oy / fh) * fh - 0.5f;
      float x0f = floorf(x), y0f = floorf(y);
      float wx = x - x0f, wy = y - y0f;
      int x0 = (int)x0f, y0 = (int)y0f;

      auto fetch = [&](int yy, int xx, float w) {
        bool valid = (xx >= 0) && (xx < wl) && (yy >= 0) && (yy < hl);
        int xc = min(max(xx, 0), wl - 1);
        int yc = min(max(yy, 0), hl - 1);
        size_t base = ((size_t)(b * LINTOK + st + yc * wl + xc) * NHEAD + h) * HDIM + ch;
        v2h vv = *(const v2h*)(value + base);
        float m = valid ? (a * w) : 0.f;
        ax += m * (float)vv[0];
        ay += m * (float)vv[1];
      };
      fetch(y0,     x0,     (1.f - wx) * (1.f - wy));
      fetch(y0,     x0 + 1, wx * (1.f - wy));
      fetch(y0 + 1, x0,     (1.f - wx) * wy);
      fetch(y0 + 1, x0 + 1, wx * wy);
    }
  }
  v2h r; r[0] = (_Float16)ax; r[1] = (_Float16)ay;
  *(v2h*)(attn + (size_t)row * DMODEL + h * HDIM + ch) = r;
}

// ---------------- launcher --------------------------------------------------
extern "C" void kernel_launch(void* const* d_in, const int* in_sizes, int n_in,
                              void* d_out, int out_size, void* d_ws, size_t ws_size,
                              hipStream_t stream) {
  const float* query  = (const float*)d_in[0];
  const float* feat   = (const float*)d_in[1];
  const float* ln_q_g = (const float*)d_in[2];
  const float* ln_q_b = (const float*)d_in[3];
  const float* ln_f_g = (const float*)d_in[4];
  const float* ln_f_b = (const float*)d_in[5];
  const float* W_off  = (const float*)d_in[6];
  const float* b_off  = (const float*)d_in[7];
  const float* W_attn = (const float*)d_in[8];
  const float* b_attn = (const float*)d_in[9];
  const float* W_val  = (const float*)d_in[10];
  const float* b_val  = (const float*)d_in[11];
  const float* W_out  = (const float*)d_in[12];
  const float* b_out  = (const float*)d_in[13];
  const float* gamma  = (const float*)d_in[14];

  char* ws = (char*)d_ws;
  _Float16* featn = (_Float16*)(ws + OFF_FEATN);
  _Float16* qn    = (_Float16*)(ws + OFF_QN);
  _Float16* value = (_Float16*)(ws + OFF_VALUE);
  _Float16* Wvalp = (_Float16*)(ws + OFF_WVALP);
  float*    Wcatf = (float*)(ws + OFF_WCATF);
  _Float16* Wcatp = (_Float16*)(ws + OFF_WCATP);
  _Float16* Woutp = (_Float16*)(ws + OFF_WOUTP);
  float*    bcat  = (float*)(ws + OFF_BCAT);
  float*    offaw = (float*)(ws + OFF_OFFAW);
  float*    awbuf = (float*)(ws + OFF_AW);
  _Float16* attn  = (_Float16*)(ws + OFF_ATTN);

  const int MROWS_F = NBATCH * LINTOK;   // 65856
  const int MROWS_Q = NBATCH * LQTOK;    // 16384

  // --- weight prep ---
  concat_w_kernel<<<(DMODEL * NCAT + 255) / 256, 256, 0, stream>>>(W_off, W_attn, Wcatf);
  concat_b_kernel<<<2, 256, 0, stream>>>(b_off, b_attn, bcat);
  pack_w_kernel<<<(24 * 48 * 32 + 255) / 256, 256, 0, stream>>>(W_val, Wvalp, DMODEL, DMODEL);
  pack_w_kernel<<<(24 * 27 * 32 + 255) / 256, 256, 0, stream>>>(Wcatf, Wcatp, DMODEL, NCAT);
  pack_w_kernel<<<(24 * 48 * 32 + 255) / 256, 256, 0, stream>>>(W_out, Woutp, DMODEL, DMODEL);

  // --- layernorms (4 waves/block, one wave per row) ---
  ln_cast_kernel<<<(MROWS_F + 3) / 4, 128, 0, stream>>>(feat, ln_f_g, ln_f_b, featn, MROWS_F);
  ln_cast_kernel<<<(MROWS_Q + 3) / 4, 128, 0, stream>>>(query, ln_q_g, ln_q_b, qn, MROWS_Q);

  // --- value projection: (65856x768)@(768x768) -> f16 ---
  {
    int stripsN = DMODEL / 64, sgN = (stripsN + 3) / 4;       // 12 strips, 3 groups
    gemm_tdm_kernel<0, 4><<<(MROWS_F >> 4) * sgN, 128, 0, stream>>>(
        featn, Wvalp, b_val, (void*)value, MROWS_F, DMODEL, DMODEL, nullptr, nullptr);
  }
  // --- offset+attn projection: (16384x768)@(768x432) -> f32 ---
  {
    int stripsN = NCAT / 48, sgN = (stripsN + 3) / 4;         // 9 strips, 3 groups
    gemm_tdm_kernel<1, 3><<<(MROWS_Q >> 4) * sgN, 128, 0, stream>>>(
        qn, Wcatp, bcat, (void*)offaw, MROWS_Q, NCAT, DMODEL, nullptr, nullptr);
  }
  // --- softmax over NL*NP ---
  softmax_kernel<<<(MROWS_Q * NHEAD + 255) / 256, 256, 0, stream>>>(offaw, awbuf);

  // --- deformable bilinear sampling ---
  {
    int waves = MROWS_Q * NHEAD;                              // 196608
    sample_kernel<<<(waves + 3) / 4, 128, 0, stream>>>(value, offaw, awbuf, attn);
  }
  // --- output projection + residual: d_out = query + gamma*(attn@W_out+b) ---
  {
    int stripsN = DMODEL / 64, sgN = (stripsN + 3) / 4;
    gemm_tdm_kernel<2, 4><<<(MROWS_Q >> 4) * sgN, 128, 0, stream>>>(
        attn, Woutp, b_out, d_out, MROWS_Q, DMODEL, DMODEL, query, gamma);
  }
}
